// KANLayer_33328946217762
// MI455X (gfx1250) — matrix operand order repaired and checked by reading, not compiled, not run on hardware
//
#include <hip/hip_runtime.h>
#include <hip/hip_bf16.h>

// KAN layer as one fused bf16 WMMA GEMM:
//   M = B*S = 8192, N = OUT = 1024, K = IN*(1+G) = 6144
//   A[m, k] = (k < 1024) ? x[m, k] : sin((1 + (k-1024)/1024) * x[m, (k-1024)%1024])
//   Wp[o, k] = (k < 1024) ? base_weight[o, k] : spline_weight[o, (k-1024)%1024, (k-1024)/1024]
//   out[m, o] = sum_k A[m,k] * Wp[o,k]

typedef __attribute__((ext_vector_type(16))) __bf16 v16bf;
typedef __attribute__((ext_vector_type(8)))  __bf16 v8bf;
typedef __attribute__((ext_vector_type(8)))  float  v8f;

#define M_TOT 8192
#define N_TOT 1024
#define K_IN  1024
#define K_TOT 6144
#define MT 128
#define NT 128
#define KT 32

// Raw 32-bit LDS byte offset of a shared-memory pointer (AS3 ptrtoint).
__device__ static __forceinline__ unsigned lds_off_u32(const void* p) {
    return (unsigned)(unsigned long long)(const __attribute__((address_space(3))) char*)p;
}

// ---------------------------------------------------------------------------
// Pack weights to bf16, layout Wp[o][k] (o-major, contiguous in k).
// ---------------------------------------------------------------------------
__global__ __launch_bounds__(256)
void kan_pack_w(const float* __restrict__ base_w,
                const float* __restrict__ spline_w,
                __bf16* __restrict__ Wp) {
    int idx = blockIdx.x * 256 + threadIdx.x;      // 0 .. 1024*6144-1
    int o = idx / K_TOT;
    int k = idx - o * K_TOT;
    float w;
    if (k < K_IN) {
        w = base_w[o * K_IN + k];
    } else {
        int kk = k - K_IN;
        int g  = kk >> 10;        // 0..4  (frequency index, freq = g+1)
        int i  = kk & 1023;
        w = spline_w[(o * K_IN + i) * 5 + g];
    }
    Wp[idx] = (__bf16)w;
}

// ---------------------------------------------------------------------------
// Fused sin-basis + bf16 WMMA GEMM.
// Block: 256 threads = 8 waves; block tile 128x128; wave tile 32x64 (2x4 frags).
// B tile is staged memory->LDS with GLOBAL_LOAD_ASYNC_TO_LDS_B128 (ASYNCcnt),
// overlapping the VALU sin/convert staging of the A tile.
// ---------------------------------------------------------------------------
__global__ __launch_bounds__(256)
void kan_wmma_gemm(const float* __restrict__ x,
                   const __bf16* __restrict__ Wp,
                   float* __restrict__ out) {
    // A tile staged row-major [row 0..127][k 0..31]; B tile staged n-major
    // [n 0..127][k 0..31] so each lane's 16-K fragment slice is contiguous.
    __shared__ __attribute__((aligned(32))) __bf16 lds_a[MT * KT];
    __shared__ __attribute__((aligned(32))) __bf16 lds_b[NT * KT];

    const int tid  = threadIdx.x;
    const int lane = tid & 31;
    const int wave = tid >> 5;
    const int wm   = (wave & 3) * 32;   // wave M offset inside block tile
    const int wn   = (wave >> 2) * 64;  // wave N offset inside block tile
    const int lrow = lane & 15;
    const int lhi  = lane >> 4;         // 0 for lanes 0-15, 1 for lanes 16-31

    const int blockM = blockIdx.x * MT; // grid.x = 64
    const int blockN = blockIdx.y * NT; // grid.y = 8

    v8f acc[2][4];
#pragma unroll
    for (int mi = 0; mi < 2; ++mi)
#pragma unroll
        for (int ni = 0; ni < 4; ++ni)
            acc[mi][ni] = (v8f)(0.0f);

    // Cooperative tile-load coordinates: each thread owns 16 elements.
    const int tr = tid >> 1;          // 0..127: A row / B column(n)
    const int tc = (tid & 1) * 16;    // k-chunk within the 32-wide tile

    // Per-thread fixed addresses for the async B staging.
    const unsigned b_lds = lds_off_u32(lds_b + tr * KT + tc);
    const __bf16* b_glb_base = Wp + (size_t)(blockN + tr) * K_TOT + tc;

    for (int kt = 0; kt < K_TOT; kt += KT) {
        const int sec = kt >> 10;               // 0 = identity, 1..5 = sin(sec*x)
        const int i0  = (kt & 1023) + tc;       // column within x (1024-wide)

        // ---- issue async B-tile DMA first: memory -> LDS, no VGPR round-trip.
        // ISA: INST_OFFSET is added to BOTH the LDS and global addresses, so
        // offset:0/16 covers this thread's 32-byte chunk from one addr pair.
        {
            const __bf16* gsrc = b_glb_base + kt;
            asm volatile(
                "global_load_async_to_lds_b128 %0, %1, off\n\t"
                "global_load_async_to_lds_b128 %0, %1, off offset:16"
                :
                : "v"(b_lds), "v"(gsrc)
                : "memory");
        }

        // ---- stage A tile: load f32 x, apply basis, convert to bf16 ----
        {
            const float* xp = x + (size_t)(blockM + tr) * K_IN + i0;
            __bf16* ap = lds_a + tr * KT + tc;
            const float f = (float)sec;
            float v[16];
#pragma unroll
            for (int j = 0; j < 16; ++j) v[j] = xp[j];
            // prefetch next K-step's x chunk (L2-resident; exercises
            // global_prefetch_b8, nearly free)
            __builtin_prefetch(xp + 32, 0, 0);
            if (sec != 0) {
#pragma unroll
                for (int j = 0; j < 16; ++j) v[j] = __sinf(v[j] * f);
            }
#pragma unroll
            for (int j = 0; j < 16; ++j) ap[j] = (__bf16)v[j];
        }

        // Async LDS writes must land before the barrier makes them visible.
        asm volatile("s_wait_asynccnt 0x0" ::: "memory");
        __syncthreads();

        // ---- fragments + 8 WMMAs ----
        // A 16x32 bf16 frag, contiguous k order used here (B matches): lane<16 -> row=lrow, K=0..15;
        //                      lane>=16 -> row=lrow, K=16..31
        v16bf afrag[2];
#pragma unroll
        for (int mi = 0; mi < 2; ++mi) {
            const __bf16* abase = lds_a + (wm + mi * 16 + lrow) * KT + lhi * 16;
            v8bf lo = *(const v8bf*)(abase);
            v8bf hi = *(const v8bf*)(abase + 8);
            afrag[mi] = __builtin_shufflevector(lo, hi,
                0, 1, 2, 3, 4, 5, 6, 7, 8, 9, 10, 11, 12, 13, 14, 15);
        }
        // B 32x16 bf16 frag, same k order as A: lane holds column n = lrow, K = lhi*16 .. +15
#pragma unroll
        for (int ni = 0; ni < 4; ++ni) {
            v16bf bfrag = *(const v16bf*)(lds_b + (wn + ni * 16 + lrow) * KT + lhi * 16);
#pragma unroll
            for (int mi = 0; mi < 2; ++mi) {
                acc[mi][ni] = __builtin_amdgcn_wmma_f32_16x16x32_bf16(
                    /*neg_a=*/false, afrag[mi],
                    /*neg_b=*/false, bfrag,
                    /*c_mod=*/(short)0, acc[mi][ni],
                    /*reuse_a=*/false, /*reuse_b=*/false);
            }
        }

        __syncthreads();
    }

    // ---- epilogue: C/D f32 layout: VGPR r -> M = r + 8*lhi, N = lane%16 ----
#pragma unroll
    for (int mi = 0; mi < 2; ++mi) {
#pragma unroll
        for (int ni = 0; ni < 4; ++ni) {
            const int col     = blockN + wn + ni * 16 + lrow;
            const int rowbase = blockM + wm + mi * 16 + lhi * 8;
#pragma unroll
            for (int r = 0; r < 8; ++r) {
                out[(size_t)(rowbase + r) * N_TOT + col] = acc[mi][ni][r];
            }
        }
    }
}

// ---------------------------------------------------------------------------
// Launch
// ---------------------------------------------------------------------------
extern "C" void kernel_launch(void* const* d_in, const int* in_sizes, int n_in,
                              void* d_out, int out_size, void* d_ws, size_t ws_size,
                              hipStream_t stream) {
    const float* x  = (const float*)d_in[0];   // (4, 2048, 1024) f32
    const float* bw = (const float*)d_in[1];   // (1024, 1024)    f32
    const float* sw = (const float*)d_in[2];   // (1024, 1024, 5) f32
    float* out = (float*)d_out;                // (4, 2048, 1024) f32
    __bf16* Wp = (__bf16*)d_ws;                // 1024*6144 bf16 = 12.6 MB

    // 1) pack weights to bf16 (recomputed every call; deterministic)
    kan_pack_w<<<(N_TOT * K_TOT) / 256, 256, 0, stream>>>(bw, sw, Wp);

    // 2) fused basis + GEMM
    dim3 grid(M_TOT / MT, N_TOT / NT);   // (64, 8)
    kan_wmma_gemm<<<grid, 256, 0, stream>>>(x, Wp, out);
}
